// TargetResponseLoss_9259949490739
// MI455X (gfx1250) — matrix operand order repaired and checked by reading, not compiled, not run on hardware
//
#include <hip/hip_runtime.h>

typedef __attribute__((ext_vector_type(16))) _Float16 v16h;
typedef __attribute__((ext_vector_type(8)))  float    v8f;

#define NSEQ 16

__device__ __forceinline__ float hat_cdf(float t) {
  t = fminf(1.f, fmaxf(-1.f, t));
  float r0 = 0.5f * (t + 1.f) * (t + 1.f);
  float r1 = 1.f - 0.5f * (1.f - t) * (1.f - t);
  return (t < 0.f) ? r0 : r1;
}

__global__ void zero_out_kernel(float* out) {
  if (threadIdx.x == 0) out[0] = 0.f;
}

__global__ void zero_sums_kernel(float* sums) {
  if (threadIdx.x == 0) { sums[0] = 0.f; sums[1] = 0.f; }
}

// Per-sequence separable PrRoI weights: wx (psz,W), wy (psz,H), inverse area.
__global__ void weights_kernel(const float* __restrict__ bb,
                               float* __restrict__ wx, float* __restrict__ wy,
                               float* __restrict__ inva,
                               int psz, int H, int W, float scale) {
  int b = blockIdx.x;
  float x1 = bb[b * 4 + 0] * scale;
  float y1 = bb[b * 4 + 1] * scale;
  float x2 = (bb[b * 4 + 0] + bb[b * 4 + 2]) * scale;
  float y2 = (bb[b * 4 + 1] + bb[b * 4 + 3]) * scale;
  float bw = (x2 - x1) / (float)psz;
  float bh = (y2 - y1) / (float)psz;
  for (int idx = threadIdx.x; idx < psz * W; idx += blockDim.x) {
    int p = idx / W, g = idx % W;
    wx[b * psz * W + idx] =
        hat_cdf(x1 + (p + 1) * bw - (float)g) - hat_cdf(x1 + p * bw - (float)g);
  }
  for (int idx = threadIdx.x; idx < psz * H; idx += blockDim.x) {
    int p = idx / H, g = idx % H;
    wy[b * psz * H + idx] =
        hat_cdf(y1 + (p + 1) * bh - (float)g) - hat_cdf(y1 + p * bh - (float)g);
  }
  if (threadIdx.x == 0) {
    float area = fmaxf(bw, 0.f) * fmaxf(bh, 0.f);
    inva[b] = (area > 0.f) ? 1.f / fmaxf(area, 1e-12f) : 0.f;
  }
}

// PrRoI pool: patch[b,c,p,q] = invarea * wy(p,:) . feat(b,c) . wx(q,:)^T
// One block per (b,c); both ref_t and ref_s share the weights.
__global__ void prroi_kernel(const float* __restrict__ featT,
                             const float* __restrict__ featS,
                             const float* __restrict__ wx,
                             const float* __restrict__ wy,
                             const float* __restrict__ inva,
                             _Float16* __restrict__ pT, _Float16* __restrict__ pS,
                             int C, int H, int W, int psz) {
  __shared__ float tT[29 * 112];
  __shared__ float tS[29 * 112];
  int bc = blockIdx.x;
  int b = bc / C;
  const float* fT = featT + (size_t)bc * H * W;
  const float* fS = featS + (size_t)bc * H * W;
  const float* wyb = wy + b * psz * H;
  const float* wxb = wx + b * psz * W;
  for (int idx = threadIdx.x; idx < psz * W; idx += blockDim.x) {
    int p = idx / W, w = idx % W;
    float sT = 0.f, sS = 0.f;
    for (int h = 0; h < H; ++h) {
      float wv = wyb[p * H + h];
      sT += wv * fT[h * W + w];
      sS += wv * fS[h * W + w];
    }
    tT[idx] = sT;
    tS[idx] = sS;
  }
  __syncthreads();
  float ia = inva[b];
  for (int idx = threadIdx.x; idx < psz * psz; idx += blockDim.x) {
    int p = idx / psz, q = idx % psz;
    float sT = 0.f, sS = 0.f;
    for (int w = 0; w < W; ++w) {
      float wv = wxb[q * W + w];
      sT += wv * tT[p * W + w];
      sS += wv * tS[p * W + w];
    }
    size_t o = (size_t)bc * psz * psz + idx;
    pT[o] = (_Float16)(sT * ia);
    pS[o] = (_Float16)(sS * ia);
  }
}

// tq[b,p,q] = sum_c |patch[b,c,p,q]|
__global__ void tq_kernel(const _Float16* __restrict__ pT,
                          const _Float16* __restrict__ pS,
                          float* __restrict__ tqT, float* __restrict__ tqS,
                          int C, int psz, int ntq) {
  int i = blockIdx.x * blockDim.x + threadIdx.x;
  if (i >= ntq) return;
  int PP = psz * psz;
  int b = i / PP, pq = i % PP;
  const _Float16* bt = pT + (size_t)b * C * PP + pq;
  const _Float16* bs = pS + (size_t)b * C * PP + pq;
  float st = 0.f, ss = 0.f;
  for (int c = 0; c < C; ++c) {
    st += fabsf((float)bt[(size_t)c * PP]);
    ss += fabsf((float)bs[(size_t)c * PP]);
  }
  tqT[i] = st;
  tqS[i] = ss;
}

// S[b,h,w] = sum_c |feat[b,c,h,w]| for both test tensors.
__global__ void abs_sum_kernel(const float* __restrict__ tt,
                               const float* __restrict__ ts,
                               float* __restrict__ Stt, float* __restrict__ Sts,
                               int C, int HW, int N) {
  int i = blockIdx.x * blockDim.x + threadIdx.x;
  if (i >= N) return;
  int b = i / HW, r = i % HW;
  const float* pt = tt + (size_t)b * C * HW + r;
  const float* ps = ts + (size_t)b * C * HW + r;
  float st = 0.f, ss = 0.f;
  for (int c = 0; c < C; ++c) {
    st += fabsf(pt[(size_t)c * HW]);
    ss += fabsf(ps[(size_t)c * HW]);
  }
  Stt[i] = st;
  Sts[i] = ss;
}

// WMMA correlation: input-row-stationary implicit GEMM.
//   A[m][kx] = img[b,c,iy, x0+m+kx-pad]   (M = 16 output x positions, K = kx)
//   B[kx][n] = filt[b,c, ky=g*16+n, kx]   (N columns = filter rows ky)
//   D[m][n] contributes to output row y = iy - ky + pad, scattered into an
//   LDS-resident H x 16 output stripe via ds_add_f32 (conflict-free).
// B fragments are built ONCE per channel (registers, NG <= 2 groups) so the
// per-row hot loop is: stage 48 halves -> 16 LDS gathers (A) -> NG WMMAs ->
// 8*NG ds_add_f32.
template <int NG>
__global__ __launch_bounds__(32)
void corr_wmma_kernel(const float* __restrict__ img,
                      const _Float16* __restrict__ filt,
                      float* __restrict__ out,
                      int C, int H, int W, int psz, int pad) {
  __shared__ float    sOut[112 * 16];
  __shared__ _Float16 sFilt[32 * 32];
  __shared__ _Float16 sA[64];

  const int lane = threadIdx.x;
  const int hi = lane >> 4;   // half-wave select
  const int n = lane & 15;    // A: row m; B/D: column n
  const int x0 = blockIdx.x * 16;
  const int b = blockIdx.y;

  for (int idx = lane; idx < H * 16; idx += 32) sOut[idx] = 0.f;

  for (int c = 0; c < C; ++c) {
    __syncthreads();
    // Stage filter channel c as [ky][kx padded to 32] f16.
    const _Float16* fsrc = filt + ((size_t)(b * C + c)) * psz * psz;
    for (int idx = lane; idx < psz * 32; idx += 32) {
      int ky = idx >> 5, kx = idx & 31;
      sFilt[idx] = (kx < psz) ? fsrc[ky * psz + kx] : (_Float16)0.f;
    }
    __syncthreads();
    // Build B fragments once per channel: lane holds column n (= filter row
    // ky for group g), k = hi*16 + 2*i + h.
    v16h bv[NG];
    bool vcol[NG];
#pragma unroll
    for (int g = 0; g < NG; ++g) {
      int kyc = g * 16 + n;
      vcol[g] = (kyc < psz);
#pragma unroll
      for (int i = 0; i < 8; ++i) {
        int kx = hi * 16 + 2 * i;
        _Float16 e0 = sFilt[(kyc & 31) * 32 + kx + 0];
        _Float16 e1 = sFilt[(kyc & 31) * 32 + kx + 1];
        bv[g][2 * i + 0] = vcol[g] ? e0 : (_Float16)0.f;
        bv[g][2 * i + 1] = vcol[g] ? e1 : (_Float16)0.f;
      }
    }
    const float* irow0 = img + ((size_t)(b * C + c)) * H * W;
    for (int iy = 0; iy < H; ++iy) {
      __syncthreads();
      // Stage image row segment: sA[t] = img[iy, x0 - pad + t], t in [0,48)
      const float* irow = irow0 + (size_t)iy * W;
      for (int t = lane; t < 48; t += 32) {
        int ix = x0 - pad + t;
        sA[t] = (_Float16)((ix >= 0 && ix < W) ? irow[ix] : 0.f);
      }
      __syncthreads();
      // Build A fragment in contiguous k order (same as B): m = n,
      // k = hi*16 + 2*i + h; A[m][k] = sA[m + k].
      v16h a;
#pragma unroll
      for (int i = 0; i < 8; ++i) {
        int k0 = hi * 16 + 2 * i;
        a[2 * i + 0] = sA[n + k0 + 0];
        a[2 * i + 1] = sA[n + k0 + 1];
      }
#pragma unroll
      for (int g = 0; g < NG; ++g) {
        v8f acc = {};
        acc = __builtin_amdgcn_wmma_f32_16x16x32_f16(
            false, a, false, bv[g], (short)0, acc, false, false);
        // Scatter D: lane column = ky, rows m = i + hi*8.
        int kyc = g * 16 + n;
        int y = iy - kyc + pad;
        if (vcol[g] && y >= 0 && y < H) {
#pragma unroll
          for (int i = 0; i < 8; ++i)
            atomicAdd(&sOut[y * 16 + hi * 8 + i], acc[i]);
        }
      }
    }
  }
  __syncthreads();
  for (int idx = lane; idx < H * 16; idx += 32) {
    int y = idx >> 4, m = idx & 15;
    int x = x0 + m;
    if (x < W) out[((size_t)b * H + y) * W + x] = sOut[idx];
  }
}

// Global sums of wt and ws for normalization.
__global__ void sum2_kernel(const float* __restrict__ a, const float* __restrict__ b,
                            float* __restrict__ sums, int N) {
  __shared__ float ra[256], rb[256];
  int i = blockIdx.x * blockDim.x + threadIdx.x;
  float va = 0.f, vb = 0.f;
  if (i < N) { va = a[i]; vb = b[i]; }
  ra[threadIdx.x] = va;
  rb[threadIdx.x] = vb;
  __syncthreads();
  for (int s = 128; s > 0; s >>= 1) {
    if ((int)threadIdx.x < s) {
      ra[threadIdx.x] += ra[threadIdx.x + s];
      rb[threadIdx.x] += rb[threadIdx.x + s];
    }
    __syncthreads();
  }
  if (threadIdx.x == 0) {
    atomicAdd(&sums[0], ra[0]);
    atomicAdd(&sums[1], rb[0]);
  }
}

// mean((qs - qt)^2) with qt = Stt * |wt| / |sum(wt)| (sum_c|tt*wt| factorizes).
__global__ void loss_resp_kernel(const float* __restrict__ wt, const float* __restrict__ ws,
                                 const float* __restrict__ Stt, const float* __restrict__ Sts,
                                 const float* __restrict__ sums,
                                 float* __restrict__ out, int N) {
  __shared__ float red[256];
  int i = blockIdx.x * blockDim.x + threadIdx.x;
  float invT = 1.f / fabsf(sums[0]);
  float invS = 1.f / fabsf(sums[1]);
  float v = 0.f;
  if (i < N) {
    float qt = Stt[i] * fabsf(wt[i]) * invT;
    float qs = Sts[i] * fabsf(ws[i]) * invS;
    float d = qs - qt;
    v = d * d;
  }
  red[threadIdx.x] = v;
  __syncthreads();
  for (int s = 128; s > 0; s >>= 1) {
    if ((int)threadIdx.x < s) red[threadIdx.x] += red[threadIdx.x + s];
    __syncthreads();
  }
  if (threadIdx.x == 0) atomicAdd(out, red[0] / (float)N);
}

// mean((tqs - tqt)^2)
__global__ void loss_tq_kernel(const float* __restrict__ tqT, const float* __restrict__ tqS,
                               float* __restrict__ out, int N) {
  __shared__ float red[256];
  int i = blockIdx.x * blockDim.x + threadIdx.x;
  float v = 0.f;
  if (i < N) {
    float d = tqS[i] - tqT[i];
    v = d * d;
  }
  red[threadIdx.x] = v;
  __syncthreads();
  for (int s = 128; s > 0; s >>= 1) {
    if ((int)threadIdx.x < s) red[threadIdx.x] += red[threadIdx.x + s];
    __syncthreads();
  }
  if (threadIdx.x == 0) atomicAdd(out, red[0] / (float)N);
}

extern "C" void kernel_launch(void* const* d_in, const int* in_sizes, int n_in,
                              void* d_out, int out_size, void* d_ws, size_t ws_size,
                              hipStream_t stream) {
  (void)in_sizes; (void)n_in; (void)out_size; (void)ws_size;
  const float* refS[4]  = {(const float*)d_in[0],  (const float*)d_in[1],
                           (const float*)d_in[2],  (const float*)d_in[3]};
  const float* testS[4] = {(const float*)d_in[4],  (const float*)d_in[5],
                           (const float*)d_in[6],  (const float*)d_in[7]};
  const float* refT[4]  = {(const float*)d_in[8],  (const float*)d_in[9],
                           (const float*)d_in[10], (const float*)d_in[11]};
  const float* testT[4] = {(const float*)d_in[12], (const float*)d_in[13],
                           (const float*)d_in[14], (const float*)d_in[15]};
  const float* bb = (const float*)d_in[16];  // (1,16,4)
  float* out = (float*)d_out;

  // Workspace layout (max over layers, reused layer-to-layer).
  size_t off = 0;
  auto take = [&](size_t bytes) -> void* {
    void* p = (char*)d_ws + off;
    off += (bytes + 255) & ~(size_t)255;
    return p;
  };
  _Float16* pT  = (_Float16*)take((size_t)NSEQ * 64 * 29 * 29 * 2);
  _Float16* pS  = (_Float16*)take((size_t)NSEQ * 64 * 29 * 29 * 2);
  float* wtb  = (float*)take((size_t)NSEQ * 112 * 112 * 4);
  float* wsb  = (float*)take((size_t)NSEQ * 112 * 112 * 4);
  float* Stt  = (float*)take((size_t)NSEQ * 112 * 112 * 4);
  float* Sts  = (float*)take((size_t)NSEQ * 112 * 112 * 4);
  float* tqT  = (float*)take((size_t)NSEQ * 29 * 29 * 4);
  float* tqS  = (float*)take((size_t)NSEQ * 29 * 29 * 4);
  float* wxb  = (float*)take((size_t)NSEQ * 29 * 112 * 4);
  float* wyb  = (float*)take((size_t)NSEQ * 29 * 112 * 4);
  float* inva = (float*)take(64 * 4);
  float* sums = (float*)take(256);

  zero_out_kernel<<<1, 32, 0, stream>>>(out);

  static const int   Cs[4] = {64, 64, 128, 256};
  static const int   Hs[4] = {112, 56, 28, 14};
  static const int   Ps[4] = {29, 15, 9, 5};
  static const float Sc[4] = {0.5f, 0.25f, 0.125f, 0.0625f};

  for (int l = 0; l < 4; ++l) {
    const int C = Cs[l], H = Hs[l], W = Hs[l];
    const int psz = Ps[l], pad = (psz - 1) / 2;
    const int W16 = (W + 15) / 16;
    const int N = NSEQ * H * W;
    const int ntq = NSEQ * psz * psz;

    weights_kernel<<<NSEQ, 128, 0, stream>>>(bb, wxb, wyb, inva, psz, H, W, Sc[l]);
    prroi_kernel<<<NSEQ * C, 256, 0, stream>>>(refT[l], refS[l], wxb, wyb, inva,
                                               pT, pS, C, H, W, psz);
    tq_kernel<<<(ntq + 255) / 256, 256, 0, stream>>>(pT, pS, tqT, tqS, C, psz, ntq);
    abs_sum_kernel<<<(N + 255) / 256, 256, 0, stream>>>(testT[l], testS[l],
                                                        Stt, Sts, C, H * W, N);
    zero_sums_kernel<<<1, 32, 0, stream>>>(sums);

    dim3 cg(W16, NSEQ);
    if (psz > 16) {
      corr_wmma_kernel<2><<<cg, 32, 0, stream>>>(testT[l], pT, wtb, C, H, W, psz, pad);
      corr_wmma_kernel<2><<<cg, 32, 0, stream>>>(testS[l], pS, wsb, C, H, W, psz, pad);
    } else {
      corr_wmma_kernel<1><<<cg, 32, 0, stream>>>(testT[l], pT, wtb, C, H, W, psz, pad);
      corr_wmma_kernel<1><<<cg, 32, 0, stream>>>(testS[l], pS, wsb, C, H, W, psz, pad);
    }

    sum2_kernel<<<(N + 255) / 256, 256, 0, stream>>>(wtb, wsb, sums, N);
    loss_resp_kernel<<<(N + 255) / 256, 256, 0, stream>>>(wtb, wsb, Stt, Sts,
                                                          sums, out, N);
    loss_tq_kernel<<<(ntq + 255) / 256, 256, 0, stream>>>(tqT, tqS, out, ntq);
  }
}
